// AccumulatorFunction_55413668053108
// MI455X (gfx1250) — compile-verified, matrix operand order checked by reading, not run on hardware
//
#include <hip/hip_runtime.h>

#define NN 8192   // neurons
#define DD 512    // state size

typedef __bf16 bf16_t;
typedef bf16_t v16bf __attribute__((ext_vector_type(16)));
typedef bf16_t v8bf  __attribute__((ext_vector_type(8)));
typedef bf16_t v4bf  __attribute__((ext_vector_type(4)));
typedef float  v8f   __attribute__((ext_vector_type(8)));
typedef float  v4f   __attribute__((ext_vector_type(4)));

// ---------------------------------------------------------------------------
// Fragment loader for WMMA bf16 16x16x32 (A and B share the pattern):
// lane holds row/col (lane%16), K offsets {half*8+0..7, 16+half*8+0..7}
// -> two contiguous 16-byte loads from a row-major [rows, stride] bf16 matrix.
// ---------------------------------------------------------------------------
static __device__ __forceinline__ v16bf load_frag(const bf16_t* __restrict__ p,
                                                  int row, int stride, int kb,
                                                  int lane) {
  const int half = lane >> 4;
  const bf16_t* base = p + (size_t)row * stride + kb + half * 8;
  v8bf lo = *(const v8bf*)(base);
  v8bf hi = *(const v8bf*)(base + 16);
  v16bf f;
#pragma unroll
  for (int e = 0; e < 8; ++e) { f[e] = lo[e]; f[e + 8] = hi[e]; }
  return f;
}

static __device__ __forceinline__ v8f wmma_bf16(v16bf a, v16bf b, v8f c) {
  return __builtin_amdgcn_wmma_f32_16x16x32_bf16(false, a, false, b, (short)0,
                                                 c, false, false);
}

// ---------------------------------------------------------------------------
// f32 -> bf16 conversion (vectorized x4)
// ---------------------------------------------------------------------------
__global__ void __launch_bounds__(256) cvt_kernel(const float* __restrict__ in,
                                                  bf16_t* __restrict__ out,
                                                  int n4) {
  int i = blockIdx.x * blockDim.x + threadIdx.x;
  if (i < n4) {
    v4f x = ((const v4f*)in)[i];
    v4bf o;
#pragma unroll
    for (int e = 0; e < 4; ++e) o[e] = (bf16_t)x[e];
    ((v4bf*)out)[i] = o;
  }
}

// ---------------------------------------------------------------------------
// Y = S @ W^T. One wave computes a 16x64 strip (4 C tiles): each A fragment
// is loaded once per K-step and reused against 4 B fragments.
// TRANSPOSED==0: Y row-major [NN,DD] bf16 (Q, K)
// TRANSPOSED==1: Y stored [DD,NN] bf16 (Vt)
// ---------------------------------------------------------------------------
template <int TRANSPOSED>
__global__ void __launch_bounds__(128) gemm16(const bf16_t* __restrict__ A,
                                              const bf16_t* __restrict__ W,
                                              bf16_t* __restrict__ Y) {
  const int lane = threadIdx.x & 31;
  const int wave = threadIdx.x >> 5;
  const int strip = blockIdx.x * 4 + wave;    // (NN/16)*(DD/64) strips
  const int tm  = (strip >> 3) * 16;          // neuron tile base
  const int td0 = (strip & 7) * 64;           // output-dim strip base (4 tiles)
  const int col = lane & 15, half = lane >> 4;

  v8f c[4];
#pragma unroll
  for (int n = 0; n < 4; ++n) c[n] = {};

#pragma unroll 2
  for (int kb = 0; kb < DD; kb += 32) {
    v16bf a = load_frag(A, tm + col, DD, kb, lane);
    v16bf b0 = load_frag(W, td0 + 0  + col, DD, kb, lane);
    v16bf b1 = load_frag(W, td0 + 16 + col, DD, kb, lane);
    v16bf b2 = load_frag(W, td0 + 32 + col, DD, kb, lane);
    v16bf b3 = load_frag(W, td0 + 48 + col, DD, kb, lane);
    c[0] = wmma_bf16(a, b0, c[0]);
    c[1] = wmma_bf16(a, b1, c[1]);
    c[2] = wmma_bf16(a, b2, c[2]);
    c[3] = wmma_bf16(a, b3, c[3]);
  }

#pragma unroll
  for (int n = 0; n < 4; ++n) {
    const int td = td0 + n * 16;
    if (TRANSPOSED) {
      v8bf o;
#pragma unroll
      for (int r = 0; r < 8; ++r) o[r] = (bf16_t)c[n][r];
      *(v8bf*)(Y + (size_t)(td + col) * NN + tm + half * 8) = o;
    } else {
#pragma unroll
      for (int r = 0; r < 8; ++r)
        Y[(size_t)(tm + half * 8 + r) * DD + td + col] = (bf16_t)c[n][r];
    }
  }
}

// ---------------------------------------------------------------------------
// Fused masked attention with online softmax (flash style).
// One wave owns 16 destination rows and the full D=512 output slice.
// Fragment loads are batched into independent register groups so the
// scheduler can clause them and overlap with the WMMA pipeline.
// ---------------------------------------------------------------------------
__global__ void __launch_bounds__(128) flash_attn(
    const bf16_t* __restrict__ Q, const bf16_t* __restrict__ Km,
    const bf16_t* __restrict__ Vt, const unsigned char* __restrict__ adj,
    float* __restrict__ out) {
  __shared__ bf16_t pbuf[4][16 * 32];  // per-wave P-tile transpose staging
  const int lane = threadIdx.x & 31;
  const int wave = threadIdx.x >> 5;
  const int col = lane & 15, half = lane >> 4;
  const int j0 = (blockIdx.x * 4 + wave) * 16;  // destination tile base
  bf16_t* pb = pbuf[wave];

  // Resident Q A-fragments: 16 x (16x32) covering D=512
  v16bf qf[16];
#pragma unroll
  for (int kk = 0; kk < 16; ++kk) qf[kk] = load_frag(Q, j0 + col, DD, kk * 32, lane);

  v8f acc[32];  // 16x512 f32 output accumulators
#pragma unroll
  for (int dt = 0; dt < 32; ++dt) acc[dt] = {};
  float mrow[8], lrow[8];
#pragma unroll
  for (int r = 0; r < 8; ++r) { mrow[r] = -1e30f; lrow[r] = 0.0f; }

  for (int s0 = 0; s0 < NN; s0 += 32) {
    // adjacency masks first (longest latency, consumed late)
    const unsigned long long mb0 =
        *(const unsigned long long*)(adj + (size_t)(s0 + col) * NN + j0 + half * 8);
    const unsigned long long mb1 =
        *(const unsigned long long*)(adj + (size_t)(s0 + 16 + col) * NN + j0 + half * 8);

    // ---- scores: two 16x16 tiles; K frags in groups of 8 independent regs ----
    v8f sc[2];
    sc[0] = {}; sc[1] = {};
#pragma unroll
    for (int t = 0; t < 2; ++t) {
#pragma unroll
      for (int h = 0; h < 2; ++h) {
        v16bf kf[8];
#pragma unroll
        for (int u = 0; u < 8; ++u)
          kf[u] = load_frag(Km, s0 + t * 16 + col, DD, (h * 8 + u) * 32, lane);
#pragma unroll
        for (int u = 0; u < 8; ++u)
          sc[t] = wmma_bf16(qf[h * 8 + u], kf[u], sc[t]);
      }
    }

    // prefetch first V group; its latency is hidden by the softmax VALU work
    v16bf vf[2][8];
#pragma unroll
    for (int u = 0; u < 8; ++u)
      vf[0][u] = load_frag(Vt, u * 16 + col, NN, s0, lane);

    // ---- online softmax per destination row (rows span 16 lanes) ----
#pragma unroll
    for (int r = 0; r < 8; ++r) {
      float v0 = sc[0][r], v1 = sc[1][r];
      float mx = fmaxf(v0, v1);
#pragma unroll
      for (int w = 1; w < 16; w <<= 1) mx = fmaxf(mx, __shfl_xor(mx, w, 32));
      float mold = mrow[r];
      float mnew = fmaxf(mold, mx);
      float scale = __expf(mold - mnew);
      mrow[r] = mnew;
      float e0 = ((mb0 >> (8 * r)) & 0xffull) ? __expf(v0 - mnew) : 0.0f;
      float e1 = ((mb1 >> (8 * r)) & 0xffull) ? __expf(v1 - mnew) : 0.0f;
      float rs = e0 + e1;
#pragma unroll
      for (int w = 1; w < 16; w <<= 1) rs += __shfl_xor(rs, w, 32);
      lrow[r] = lrow[r] * scale + rs;
#pragma unroll
      for (int dt = 0; dt < 32; ++dt) acc[dt][r] *= scale;
      pb[(half * 8 + r) * 32 + col]      = (bf16_t)e0;
      pb[(half * 8 + r) * 32 + 16 + col] = (bf16_t)e1;
    }

    // re-read P as an A-fragment (transpose through LDS)
    v16bf pf;
    {
      const bf16_t* base = pb + col * 32 + half * 8;
      v8bf lo = *(const v8bf*)(base);
      v8bf hi = *(const v8bf*)(base + 16);
#pragma unroll
      for (int e = 0; e < 8; ++e) { pf[e] = lo[e]; pf[e + 8] = hi[e]; }
    }

    // ---- O += P @ V, double-buffered fragment groups of 8 ----
#pragma unroll
    for (int g = 0; g < 4; ++g) {
      if (g < 3) {
#pragma unroll
        for (int u = 0; u < 8; ++u)
          vf[(g + 1) & 1][u] =
              load_frag(Vt, ((g + 1) * 8 + u) * 16 + col, NN, s0, lane);
      }
#pragma unroll
      for (int u = 0; u < 8; ++u)
        acc[g * 8 + u] = wmma_bf16(pf, vf[g & 1][u], acc[g * 8 + u]);
    }
  }

  // ---- epilogue: divide by row sums; zero-predecessor rows -> 0 ----
#pragma unroll
  for (int r = 0; r < 8; ++r) {
    float inv = (lrow[r] > 0.0f) ? (1.0f / lrow[r]) : 0.0f;
#pragma unroll
    for (int dt = 0; dt < 32; ++dt)
      out[(size_t)(j0 + half * 8 + r) * DD + dt * 16 + col] = acc[dt][r] * inv;
  }
}

// ---------------------------------------------------------------------------
extern "C" void kernel_launch(void* const* d_in, const int* in_sizes, int n_in,
                              void* d_out, int out_size, void* d_ws,
                              size_t ws_size, hipStream_t stream) {
  const float* S = (const float*)d_in[0];
  const unsigned char* adj = (const unsigned char*)d_in[1];  // bool bytes
  const float* Wq = (const float*)d_in[2];
  const float* Wk = (const float*)d_in[3];
  const float* Wv = (const float*)d_in[4];
  float* out = (float*)d_out;

  char* ws = (char*)d_ws;
  size_t off = 0;
  bf16_t* Sbf = (bf16_t*)(ws + off); off += (size_t)NN * DD * 2;  // 8 MiB
  bf16_t* Wqb = (bf16_t*)(ws + off); off += (size_t)DD * DD * 2;  // 512 KiB
  bf16_t* Wkb = (bf16_t*)(ws + off); off += (size_t)DD * DD * 2;
  bf16_t* Wvb = (bf16_t*)(ws + off); off += (size_t)DD * DD * 2;
  bf16_t* Qb  = (bf16_t*)(ws + off); off += (size_t)NN * DD * 2;  // Q [N,D]
  bf16_t* Kb  = (bf16_t*)(ws + off); off += (size_t)NN * DD * 2;  // K [N,D]
  bf16_t* Vtb = (bf16_t*)(ws + off); off += (size_t)NN * DD * 2;  // V^T [D,N]
  (void)ws_size; (void)in_sizes; (void)n_in; (void)out_size;

  // 1) bf16 conversion
  {
    int n4 = NN * DD / 4;
    cvt_kernel<<<n4 / 256, 256, 0, stream>>>(S, Sbf, n4);
    n4 = DD * DD / 4;
    cvt_kernel<<<n4 / 256, 256, 0, stream>>>(Wq, Wqb, n4);
    cvt_kernel<<<n4 / 256, 256, 0, stream>>>(Wk, Wkb, n4);
    cvt_kernel<<<n4 / 256, 256, 0, stream>>>(Wv, Wvb, n4);
  }

  // 2) Q/K/V projections: (NN/16)*(DD/64)=4096 strips, 4 waves per block
  const int gemm_blocks = (NN / 16) * (DD / 64) / 4;  // 1024
  gemm16<0><<<gemm_blocks, 128, 0, stream>>>(Sbf, Wqb, Qb);
  gemm16<0><<<gemm_blocks, 128, 0, stream>>>(Sbf, Wkb, Kb);
  gemm16<1><<<gemm_blocks, 128, 0, stream>>>(Sbf, Wvb, Vtb);

  // 3) fused masked attention (512 dst tiles / 4 waves per block)
  flash_attn<<<NN / 16 / 4, 128, 0, stream>>>(Qb, Kb, Vtb, adj, out);
}
